// Vanilla_82592221102214
// MI455X (gfx1250) — compile-verified, matrix operand order checked by reading, not run on hardware
//
#include <hip/hip_runtime.h>

// ---------------------------------------------------------------------------
// Fused vanilla-RNN for MI455X (gfx1250, wave32, WMMA bf16).
//   readouts[B,T,O], hs[B,T,U] = RNN(x[B,T,I]; W_ih, W_hh, b_ih, b_hh, W_fc, b_fc)
// One workgroup per 16-row batch slice. W_ih / W_hh B-fragments live in
// VGPRs for the whole sequence; only activations go through LDS. Single
// barrier per timestep (ping-ponged xs / h_raw); readout GEMM for step t-1
// runs inside step t; x is register double-buffered. The kill_neuron mask
// (units 3,7) is applied by AND-masking the kb=0 recurrence A-fragment
// instead of keeping a separate h_masked LDS buffer.
// ---------------------------------------------------------------------------

typedef __bf16 bf16_t;
typedef bf16_t v16bf __attribute__((ext_vector_type(16)));
typedef bf16_t v8bf  __attribute__((ext_vector_type(8)));
typedef float  v8f   __attribute__((ext_vector_type(8)));
typedef float  v4f   __attribute__((ext_vector_type(4)));

constexpr int kB = 128;
constexpr int kT = 2048;
constexpr int kI = 128;
constexpr int kU = 256;
constexpr int kO = 128;

// LDS layout (bf16 element offsets). Rows padded by 8 elems (16 B).
constexpr int WFC_PITCH = kU + 8;   // 264
constexpr int XS_PITCH  = kI + 8;   // 136
constexpr int H_PITCH   = kU + 8;   // 264

constexpr int WFC_OFF  = 0;
constexpr int WFC_SZ   = kO * WFC_PITCH;              // 33792
constexpr int XS_SZ    = 16 * XS_PITCH;               // 2176
constexpr int XS0_OFF  = WFC_OFF + WFC_SZ;
constexpr int XS1_OFF  = XS0_OFF + XS_SZ;
constexpr int H_SZ     = 16 * H_PITCH;                // 4224
constexpr int HRAW0_OFF = XS1_OFF + XS_SZ;
constexpr int HRAW1_OFF = HRAW0_OFF + H_SZ;
constexpr int LDS_ELEMS = HRAW1_OFF + H_SZ;           // 46592
constexpr int LDS_BYTES = LDS_ELEMS * 2;              // 93184 B

union ABFrag { v16bf v; v8bf h[2]; unsigned u[8]; };

// Two contiguous 16-byte LDS chunks -> one 16-bit A/B operand fragment
// (lanes 0-15: K {base..+7, base+16..+23}; lanes 16-31 pre-shifted +8 via
// the caller's `hi*8`).
__device__ inline v16bf load_frag(const bf16_t* p) {
  ABFrag f;
  f.h[0] = *(const v8bf*)(p);
  f.h[1] = *(const v8bf*)(p + 16);
  return f.v;
}

// Same, but zero hidden units 3 and 7 (kill_neuron): they sit in the high
// halves of dwords 1 and 3 of the kb=0 fragment for lanes 0-15.
__device__ inline v16bf load_frag_masked(const bf16_t* p, unsigned m) {
  ABFrag f;
  f.h[0] = *(const v8bf*)(p);
  f.h[1] = *(const v8bf*)(p + 16);
  f.u[1] &= m;
  f.u[3] &= m;
  return f.v;
}

// 8 contiguous fp32 -> 8 bf16 (half a fragment), straight from global.
__device__ inline v8bf cvt8(const float* p) {
  v4f a = *(const v4f*)p;
  v4f b = *(const v4f*)(p + 4);
  v8bf r;
  r[0] = (bf16_t)a.x; r[1] = (bf16_t)a.y; r[2] = (bf16_t)a.z; r[3] = (bf16_t)a.w;
  r[4] = (bf16_t)b.x; r[5] = (bf16_t)b.y; r[6] = (bf16_t)b.z; r[7] = (bf16_t)b.w;
  return r;
}

__device__ inline v16bf make_bfrag(const float* row, int kbase) {
  ABFrag f;
  f.h[0] = cvt8(row + kbase);
  f.h[1] = cvt8(row + kbase + 16);
  return f.v;
}

__device__ inline void wmma_bf16(v16bf a, v16bf b, v8f& acc) {
  acc = __builtin_amdgcn_wmma_f32_16x16x32_bf16(false, a, false, b,
                                                (short)0, acc, false, false);
}

__global__ __launch_bounds__(512)
void rnn_fused_kernel(const float* __restrict__ x,
                      const float* __restrict__ W_ih,
                      const float* __restrict__ W_hh,
                      const float* __restrict__ b_ih,
                      const float* __restrict__ b_hh,
                      const float* __restrict__ W_fc,
                      const float* __restrict__ b_fc,
                      float* __restrict__ ro,   // [B,T,O]
                      float* __restrict__ hs)   // [B,T,U]
{
  extern __shared__ bf16_t lds[];

  const int tid  = threadIdx.x;
  const int wave = tid >> 5;
  const int lane = tid & 31;
  const int l16  = lane & 15;
  const int hi   = lane >> 4;          // K half-select for A/B fragments
  const int b0   = blockIdx.x * 16;    // batch rows [b0, b0+16)

  // ---- per-lane constants --------------------------------------------------
  const int      colU  = wave * 16 + l16;              // hidden-unit column
  const float    xbias = b_ih[colU] + b_hh[colU];
  const float    fmask = (colU == 3 || colU == 7) ? 0.0f : 1.0f;
  const unsigned maskA = (hi == 0) ? 0x0000FFFFu : 0xFFFFFFFFu;  // kill 3,7
  const int      colO  = (wave & 7) * 16 + l16;        // readout column
  const float    fbias = b_fc[colO];
  const int      mrow  = hi * 8;                       // C-matrix row base

  // ---- loop-invariant weight B-fragments, held in VGPRs for all 2048 steps
  v16bf whhB[8];                       // W_hh row colU, K = 256 (64 VGPRs)
  {
    const float* row = W_hh + (size_t)colU * kU + hi * 8;
    #pragma unroll
    for (int kb = 0; kb < 8; ++kb) whhB[kb] = make_bfrag(row, kb * 32);
  }
  v16bf wihB[4];                       // W_ih row colU, K = 128 (32 VGPRs)
  {
    const float* row = W_ih + (size_t)colU * kI + hi * 8;
    #pragma unroll
    for (int kb = 0; kb < 4; ++kb) wihB[kb] = make_bfrag(row, kb * 32);
  }

  // ---- stage W_fc into LDS as bf16 (readout is off the critical path) -----
  for (int n = tid * 4; n < kO * kU; n += 512 * 4) {
    v4f w = *(const v4f*)(W_fc + n);
    bf16_t* p = lds + WFC_OFF + (n / kU) * WFC_PITCH + (n % kU);
    p[0] = (bf16_t)w.x; p[1] = (bf16_t)w.y; p[2] = (bf16_t)w.z; p[3] = (bf16_t)w.w;
  }
  // h(-1) = 0 lives in hraw buffer 1 (parity (t-1)&1 at t=0)
  for (int n = tid; n < H_SZ; n += 512) lds[HRAW1_OFF + n] = (bf16_t)0.0f;

  const bf16_t* wfcrow = lds + WFC_OFF + colO * WFC_PITCH + hi * 8;

  // readout for a finished step `tp` (reads hraw[tp&1]); all 16 waves
  // compute (EXEC all-1s around WMMA), waves 0-7 store.
  auto do_readout = [&](int tp) {
    const bf16_t* hrow = lds + ((tp & 1) ? HRAW1_OFF : HRAW0_OFF)
                             + l16 * H_PITCH + hi * 8;
    v8f yacc;
    #pragma unroll
    for (int j = 0; j < 8; ++j) yacc[j] = fbias;
    #pragma unroll
    for (int kb = 0; kb < 8; ++kb) {
      v16bf a = load_frag(hrow   + kb * 32);
      v16bf b = load_frag(wfcrow + kb * 32);
      wmma_bf16(a, b, yacc);
    }
    if (wave < 8) {
      #pragma unroll
      for (int j = 0; j < 8; ++j)
        ro[((size_t)(b0 + mrow + j) * kT + tp) * kO + colO] = yacc[j];
    }
  };

  // register double-buffer for this thread's 16-byte slice of x[:, t, :]
  const int sn = tid * 4;                 // 512 threads x float4 = 16x128
  const int sm = sn >> 7;                 // batch row within slice
  const int sc = sn & (kI - 1);           // input feature
  const float* xgp = x + ((size_t)(b0 + sm) * kT) * kI + sc;   // t = 0
  v4f xreg = *(const v4f*)xgp;

  __syncthreads();   // W_fc + h(-1) ready

  for (int t = 0; t < kT; ++t) {
    // ---- publish x(t) (bf16) into xs[t&1] ---------------------------------
    {
      bf16_t* p = lds + ((t & 1) ? XS1_OFF : XS0_OFF) + sm * XS_PITCH + sc;
      p[0] = (bf16_t)xreg.x; p[1] = (bf16_t)xreg.y;
      p[2] = (bf16_t)xreg.z; p[3] = (bf16_t)xreg.w;
    }
    __syncthreads();   // xs(t) and hraw[(t-1)&1] visible

    // kick off next step's x load immediately: HBM latency hides under WMMAs
    if (t + 1 < kT) {
      xgp += kI;
      xreg = *(const v4f*)xgp;
      __builtin_prefetch(xgp + kI, 0, 3);   // global_prefetch_b8 for t+2
    }

    // independent work first: readout GEMM of the previous step
    if (t > 0) do_readout(t - 1);

    const bf16_t* xsrow = lds + ((t & 1) ? XS1_OFF : XS0_OFF)
                              + l16 * XS_PITCH + hi * 8;
    const bf16_t* hprow = lds + ((t & 1) ? HRAW1_OFF : HRAW0_OFF)
                              + l16 * H_PITCH + hi * 8;    // hraw[t&1] = h(t) write buffer; unused
    bf16_t* hrw = lds + ((t & 1) ? HRAW0_OFF : HRAW1_OFF); // h_raw(t)  [note]
    // parity: write buffer (t&1)^1 == ((t+1)&1); read buffer (t-1)&1 == (t+1)&1?
    // No: (t-1)&1 == (t+1)&1, so read and write would collide -- read must be
    // the OPPOSITE parity of write.  h(t) goes to hraw[t&1]:
    hrw = lds + ((t & 1) ? HRAW1_OFF : HRAW0_OFF);
    const bf16_t* hrd = lds + ((t & 1) ? HRAW0_OFF : HRAW1_OFF)
                            + l16 * H_PITCH + hi * 8;      // h_raw(t-1)
    (void)hprow;

    v8f acc;
    #pragma unroll
    for (int j = 0; j < 8; ++j) acc[j] = xbias;

    // x-projection: K = 128; B from registers, A from LDS (2 ds_loads/WMMA)
    #pragma unroll
    for (int kb = 0; kb < 4; ++kb)
      wmma_bf16(load_frag(xsrow + kb * 32), wihB[kb], acc);

    // recurrence: K = 256; B from registers, A from LDS; kb=0 fragment is
    // AND-masked to zero units 3 and 7 (the carried state is the masked one)
    wmma_bf16(load_frag_masked(hrd, maskA), whhB[0], acc);
    #pragma unroll
    for (int kb = 1; kb < 8; ++kb)
      wmma_bf16(load_frag(hrd + kb * 32), whhB[kb], acc);

    // tanh; publish h_raw(t) to LDS, stream masked h to HBM (fp32)
    #pragma unroll
    for (int j = 0; j < 8; ++j) {
      const int   m  = mrow + j;
      const float hr = tanhf(acc[j]);
      hrw[m * H_PITCH + colU] = (bf16_t)hr;
      hs[((size_t)(b0 + m) * kT + t) * kU + colU] = hr * fmask;
    }
  }

  __syncthreads();            // final hraw visible
  do_readout(kT - 1);
}

extern "C" void kernel_launch(void* const* d_in, const int* in_sizes, int n_in,
                              void* d_out, int out_size, void* d_ws, size_t ws_size,
                              hipStream_t stream) {
  (void)in_sizes; (void)n_in; (void)out_size; (void)d_ws; (void)ws_size;
  const float* x    = (const float*)d_in[0];
  const float* W_ih = (const float*)d_in[1];
  const float* W_hh = (const float*)d_in[2];
  const float* b_ih = (const float*)d_in[3];
  const float* b_hh = (const float*)d_in[4];
  const float* W_fc = (const float*)d_in[5];
  const float* b_fc = (const float*)d_in[6];
  float* ro = (float*)d_out;                      // readouts [B,T,O] first
  float* hs = ro + (size_t)kB * kT * kO;          // then hs  [B,T,U]

  dim3 grid(kB / 16), block(512);
  rnn_fused_kernel<<<grid, block, LDS_BYTES, stream>>>(
      x, W_ih, W_hh, b_ih, b_hh, W_fc, b_fc, ro, hs);
}
